// Tag_Transformer_82291573392057
// MI455X (gfx1250) — compile-verified, matrix operand order checked by reading, not run on hardware
//
#include <hip/hip_runtime.h>
#include <hip/hip_bf16.h>

typedef __bf16 bf16_t;
typedef __attribute__((ext_vector_type(16))) __bf16 v16bf;
typedef __attribute__((ext_vector_type(8)))  __bf16 v8bf;
typedef __attribute__((ext_vector_type(4)))  __bf16 v4bf;
typedef __attribute__((ext_vector_type(8)))  float  v8f;

#define LAYERS 4
#define BATCH  64
#define TCACHE 511
#define DMODEL 1024
#define NHEAD  16
#define DHEAD  64
#define DFFN   4096
#define SMEMT  256
#define ATT_SCALE 0.125f   /* 1/sqrt(64) */

#define BM 128
#define BN 128
#define BK 64
#define LDSS 72            /* bf16 LDS row stride (64 + 8 pad, 144B = 8B-aligned) */

// ---------------------------------------------------------------------------
// Load one 16x32 bf16 fragment (A-style: lane -> row, VGPRs stride K) from
// LDS at column offset kOff.  Layout per ISA 7.12.2.
// ---------------------------------------------------------------------------
__device__ inline v16bf load_frag(const bf16_t* base, int lane, int kOff) {
  v16bf f;
  const int r  = lane & 15;
  const int kh = (lane >> 4) << 3;
  const bf16_t* p = base + r * LDSS + kOff;
#pragma unroll
  for (int v = 0; v < 8; ++v) {
    const int k = ((v >> 2) << 4) + kh + ((v & 3) << 1);
    f[2 * v]     = p[k];
    f[2 * v + 1] = p[k + 1];
  }
  return f;
}

// ---------------------------------------------------------------------------
// C = concat(A0[0:M0], A1)[M rows] @ W^T + bias.  A row-major stride K,
// W is (N,K) row-major.  128x128 block tile, BK=64, 8 waves each computing a
// 32x64 patch via 16 v_wmma_f32_16x16x32_bf16 per K-step.  M guarded
// (clamped loads, predicated stores); N multiple of 128, K multiple of 64.
// Optional ReLU; output f32 (Cf) or bf16 (Cb).
// ---------------------------------------------------------------------------
__global__ __launch_bounds__(256)
void gemm_wmma_bf16(const float* __restrict__ A0, const float* __restrict__ A1,
                    long M0, long M, const float* __restrict__ W,
                    const float* __restrict__ bias, int K,
                    float* __restrict__ Cf, bf16_t* __restrict__ Cb,
                    int ldc, int relu)
{
  __shared__ bf16_t As[BM * LDSS];
  __shared__ bf16_t Ws[BN * LDSS];

  const int tid  = threadIdx.x;
  const int lane = tid & 31;
  const int w    = tid >> 5;       // 8 waves
  const int wm   = w >> 1;         // 0..3 -> 32-row strip
  const int wn   = w & 1;          // 0..1 -> 64-col strip

  v8f acc[2][4];
#pragma unroll
  for (int mi = 0; mi < 2; ++mi)
#pragma unroll
    for (int ni = 0; ni < 4; ++ni)
      acc[mi][ni] = (v8f){0.f, 0.f, 0.f, 0.f, 0.f, 0.f, 0.f, 0.f};

  const long rowBase = (long)blockIdx.y * BM;
  const long colBase = (long)blockIdx.x * BN;

  for (int k0 = 0; k0 < K; k0 += BK) {
    if (k0 + BK < K) {  // gfx1250 global_prefetch_b8 of the next tiles
      __builtin_prefetch(W + (colBase + (tid >> 1)) * (long)K + k0 + BK +
                             (tid & 1) * 32, 0, 1);
      long pr = rowBase + (tid >> 1);
      if (pr >= M) pr = M - 1;
      const float* psrc = (pr < M0) ? (A0 + pr * (long)K)
                                    : (A1 + (pr - M0) * (long)K);
      __builtin_prefetch(psrc + k0 + BK + (tid & 1) * 32, 0, 1);
    }
    // A tile: 128 x 64 f32 -> bf16 (float4 loads, 4xbf16 packed LDS stores)
#pragma unroll
    for (int i = 0; i < 8; ++i) {
      const int slot = tid + i * 256;          // 2048 float4 slots
      const int r = slot >> 4, kg = slot & 15;
      long gr = rowBase + r;
      if (gr >= M) gr = M - 1;
      const float* src = (gr < M0) ? (A0 + gr * (long)K)
                                   : (A1 + (gr - M0) * (long)K);
      const float4 f = *(const float4*)(src + k0 + kg * 4);
      v4bf h = {(bf16_t)f.x, (bf16_t)f.y, (bf16_t)f.z, (bf16_t)f.w};
      *(v4bf*)(As + r * LDSS + kg * 4) = h;
    }
    // W tile: 128 x 64
#pragma unroll
    for (int i = 0; i < 8; ++i) {
      const int slot = tid + i * 256;
      const int r = slot >> 4, kg = slot & 15;
      const float4 f =
          *(const float4*)(W + (colBase + r) * (long)K + k0 + kg * 4);
      v4bf h = {(bf16_t)f.x, (bf16_t)f.y, (bf16_t)f.z, (bf16_t)f.w};
      *(v4bf*)(Ws + r * LDSS + kg * 4) = h;
    }
    __syncthreads();

#pragma unroll
    for (int kk = 0; kk < 2; ++kk) {           // two 32-K substeps
      const int kOff = kk * 32;
      v16bf a0 = load_frag(As + (wm * 32) * LDSS,      lane, kOff);
      v16bf a1 = load_frag(As + (wm * 32 + 16) * LDSS, lane, kOff);
      v16bf b0 = load_frag(Ws + (wn * 64) * LDSS,      lane, kOff);
      v16bf b1 = load_frag(Ws + (wn * 64 + 16) * LDSS, lane, kOff);
      v16bf b2 = load_frag(Ws + (wn * 64 + 32) * LDSS, lane, kOff);
      v16bf b3 = load_frag(Ws + (wn * 64 + 48) * LDSS, lane, kOff);
      acc[0][0] = __builtin_amdgcn_wmma_f32_16x16x32_bf16(false, a0, false, b0, (short)0, acc[0][0], false, false);
      acc[0][1] = __builtin_amdgcn_wmma_f32_16x16x32_bf16(false, a0, false, b1, (short)0, acc[0][1], false, false);
      acc[0][2] = __builtin_amdgcn_wmma_f32_16x16x32_bf16(false, a0, false, b2, (short)0, acc[0][2], false, false);
      acc[0][3] = __builtin_amdgcn_wmma_f32_16x16x32_bf16(false, a0, false, b3, (short)0, acc[0][3], false, false);
      acc[1][0] = __builtin_amdgcn_wmma_f32_16x16x32_bf16(false, a1, false, b0, (short)0, acc[1][0], false, false);
      acc[1][1] = __builtin_amdgcn_wmma_f32_16x16x32_bf16(false, a1, false, b1, (short)0, acc[1][1], false, false);
      acc[1][2] = __builtin_amdgcn_wmma_f32_16x16x32_bf16(false, a1, false, b2, (short)0, acc[1][2], false, false);
      acc[1][3] = __builtin_amdgcn_wmma_f32_16x16x32_bf16(false, a1, false, b3, (short)0, acc[1][3], false, false);
    }
    __syncthreads();
  }

  // C/D layout: n = lane%16, m = v + 8*(lane/16)
  const int mrow = (lane >> 4) << 3;
  const int ncol = lane & 15;
#pragma unroll
  for (int mi = 0; mi < 2; ++mi) {
#pragma unroll
    for (int v = 0; v < 8; ++v) {
      const long row = rowBase + wm * 32 + mi * 16 + mrow + v;
      if (row < M) {
#pragma unroll
        for (int ni = 0; ni < 4; ++ni) {
          const long col = colBase + wn * 64 + ni * 16 + ncol;
          float val = acc[mi][ni][v] + bias[col];
          if (relu) val = fmaxf(val, 0.f);
          if (Cb) Cb[row * (long)ldc + col] = (bf16_t)val;
          else    Cf[row * (long)ldc + col] = val;
        }
      }
    }
  }
}

// ---------------------------------------------------------------------------
// Attention over f32 K/V in (b,h,t,dh) layout, optional appended last row
// (layer-0 new token k/v, row stride newStride).  One block per (b,h).
// ---------------------------------------------------------------------------
__global__ __launch_bounds__(256)
void attn_f32(const float* __restrict__ Q, int qStride,
              const float* __restrict__ Kb, const float* __restrict__ Vb,
              int Tkv, const float* __restrict__ knew,
              const float* __restrict__ vnew, int newStride,
              float* __restrict__ ctx)
{
  __shared__ float qs[DHEAD];
  __shared__ float sc[512];
  __shared__ float red[256];

  const int tid = threadIdx.x;
  const int b = blockIdx.x / NHEAD;
  const int h = blockIdx.x % NHEAD;
  const int Tlen = Tkv + (knew ? 1 : 0);

  if (tid < DHEAD) qs[tid] = Q[(long)b * qStride + h * DHEAD + tid];
  __syncthreads();

  float lmax = -3.0e38f;
  for (int t = tid; t < Tlen; t += 256) {
    const float* kp = (t < Tkv)
        ? (Kb + (((long)(b * NHEAD + h)) * Tkv + t) * DHEAD)
        : (knew + (long)b * newStride + h * DHEAD);
    const float4* kp4 = (const float4*)kp;
    float d = 0.f;
#pragma unroll
    for (int j = 0; j < DHEAD / 4; ++j) {
      const float4 kk = kp4[j];
      d += qs[4 * j] * kk.x + qs[4 * j + 1] * kk.y +
           qs[4 * j + 2] * kk.z + qs[4 * j + 3] * kk.w;
    }
    d *= ATT_SCALE;
    sc[t] = d;
    lmax = fmaxf(lmax, d);
  }
  red[tid] = lmax; __syncthreads();
  for (int st = 128; st > 0; st >>= 1) {
    if (tid < st) red[tid] = fmaxf(red[tid], red[tid + st]);
    __syncthreads();
  }
  const float mx = red[0]; __syncthreads();

  float lsum = 0.f;
  for (int t = tid; t < Tlen; t += 256) {
    const float e = __expf(sc[t] - mx);
    sc[t] = e;
    lsum += e;
  }
  red[tid] = lsum; __syncthreads();
  for (int st = 128; st > 0; st >>= 1) {
    if (tid < st) red[tid] += red[tid + st];
    __syncthreads();
  }
  const float inv = 1.f / red[0]; __syncthreads();

  const int dh = tid >> 2, c = tid & 3;
  const int tpc = (Tlen + 3) >> 2;
  const int t0 = c * tpc;
  int t1 = t0 + tpc; if (t1 > Tlen) t1 = Tlen;
  float acc = 0.f;
  for (int t = t0; t < t1; ++t) {
    const float* vp = (t < Tkv)
        ? (Vb + (((long)(b * NHEAD + h)) * Tkv + t) * DHEAD)
        : (vnew + (long)b * newStride + h * DHEAD);
    acc += sc[t] * vp[dh];
  }
  red[tid] = acc; __syncthreads();
  if (c == 0)
    ctx[(long)b * DMODEL + h * DHEAD + dh] =
        (red[dh * 4] + red[dh * 4 + 1] + red[dh * 4 + 2] + red[dh * 4 + 3]) * inv;
}

// ---------------------------------------------------------------------------
// Attention over bf16 KV in (t,b,[K|V] h dh) layout produced by the KV GEMM:
// row (t*B+b) has 2048 cols: [0,1024) = K, [1024,2048) = V.  T = 512.
// ---------------------------------------------------------------------------
__global__ __launch_bounds__(256)
void attn_kv_bf16(const float* __restrict__ Q, const bf16_t* __restrict__ KV,
                  float* __restrict__ ctx)
{
  __shared__ float qs[DHEAD];
  __shared__ float sc[512];
  __shared__ float red[256];

  const int tid = threadIdx.x;
  const int b = blockIdx.x / NHEAD;
  const int h = blockIdx.x % NHEAD;
  const int Tlen = TCACHE + 1;  // 512

  if (tid < DHEAD) qs[tid] = Q[(long)b * DMODEL + h * DHEAD + tid];
  __syncthreads();

  float lmax = -3.0e38f;
  for (int t = tid; t < Tlen; t += 256) {
    const v8bf* kp8 = (const v8bf*)(KV +
        ((long)(t * BATCH + b)) * (2 * DMODEL) + h * DHEAD);
    float d = 0.f;
#pragma unroll
    for (int j = 0; j < DHEAD / 8; ++j) {
      const v8bf kk = kp8[j];
#pragma unroll
      for (int e = 0; e < 8; ++e) d += qs[8 * j + e] * (float)kk[e];
    }
    d *= ATT_SCALE;
    sc[t] = d;
    lmax = fmaxf(lmax, d);
  }
  red[tid] = lmax; __syncthreads();
  for (int st = 128; st > 0; st >>= 1) {
    if (tid < st) red[tid] = fmaxf(red[tid], red[tid + st]);
    __syncthreads();
  }
  const float mx = red[0]; __syncthreads();

  float lsum = 0.f;
  for (int t = tid; t < Tlen; t += 256) {
    const float e = __expf(sc[t] - mx);
    sc[t] = e;
    lsum += e;
  }
  red[tid] = lsum; __syncthreads();
  for (int st = 128; st > 0; st >>= 1) {
    if (tid < st) red[tid] += red[tid + st];
    __syncthreads();
  }
  const float inv = 1.f / red[0]; __syncthreads();

  const int dh = tid >> 2, c = tid & 3;
  const int tpc = Tlen >> 2;
  float acc = 0.f;
  for (int t = c * tpc; t < (c + 1) * tpc; ++t) {
    const bf16_t* vp =
        KV + ((long)(t * BATCH + b)) * (2 * DMODEL) + DMODEL + h * DHEAD;
    acc += sc[t] * (float)vp[dh];
  }
  red[tid] = acc; __syncthreads();
  if (c == 0)
    ctx[(long)b * DMODEL + h * DHEAD + dh] =
        (red[dh * 4] + red[dh * 4 + 1] + red[dh * 4 + 2] + red[dh * 4 + 3]) * inv;
}

// ---------------------------------------------------------------------------
// y = LayerNorm(a + d) * g + beta   over D=1024, one block per row (B=64).
// Each thread owns exactly one float4 (256 * 4 = 1024).
// ---------------------------------------------------------------------------
__global__ __launch_bounds__(256)
void ln_res(const float* __restrict__ a, const float* __restrict__ d,
            const float* __restrict__ g, const float* __restrict__ be,
            float* __restrict__ y)
{
  __shared__ float r1[256], r2[256];
  const int row = blockIdx.x, tid = threadIdx.x;
  const float4 av = ((const float4*)(a + (long)row * DMODEL))[tid];
  const float4 dv = ((const float4*)(d + (long)row * DMODEL))[tid];
  float v[4] = {av.x + dv.x, av.y + dv.y, av.z + dv.z, av.w + dv.w};
  float s = v[0] + v[1] + v[2] + v[3];
  float sq = v[0] * v[0] + v[1] * v[1] + v[2] * v[2] + v[3] * v[3];
  r1[tid] = s; r2[tid] = sq; __syncthreads();
  for (int st = 128; st > 0; st >>= 1) {
    if (tid < st) { r1[tid] += r1[tid + st]; r2[tid] += r2[tid + st]; }
    __syncthreads();
  }
  const float mean = r1[0] * (1.f / DMODEL);
  const float var  = r2[0] * (1.f / DMODEL) - mean * mean;
  const float rstd = rsqrtf(var + 1e-5f);
  const float4 gv  = ((const float4*)g)[tid];
  const float4 bv  = ((const float4*)be)[tid];
  float4 ov;
  ov.x = (v[0] - mean) * rstd * gv.x + bv.x;
  ov.y = (v[1] - mean) * rstd * gv.y + bv.y;
  ov.z = (v[2] - mean) * rstd * gv.z + bv.z;
  ov.w = (v[3] - mean) * rstd * gv.w + bv.w;
  ((float4*)(y + (long)row * DMODEL))[tid] = ov;
}

// ---------------------------------------------------------------------------
extern "C" void kernel_launch(void* const* d_in, const int* in_sizes, int n_in,
                              void* d_out, int out_size, void* d_ws,
                              size_t ws_size, hipStream_t stream)
{
  const float* tgt_last = (const float*)d_in[0];
  const float* cache    = (const float*)d_in[1];
  const float* K0       = (const float*)d_in[2];
  const float* V0       = (const float*)d_in[3];
  const float* K_mem    = (const float*)d_in[4];
  const float* V_mem    = (const float*)d_in[5];
  const float* sa_W     = (const float*)d_in[6];
  const float* sa_b     = (const float*)d_in[7];
  const float* sa_oW    = (const float*)d_in[8];
  const float* sa_ob    = (const float*)d_in[9];
  const float* ca_qW    = (const float*)d_in[10];
  const float* ca_qb    = (const float*)d_in[11];
  const float* ca_oW    = (const float*)d_in[12];
  const float* ca_ob    = (const float*)d_in[13];
  const float* W1       = (const float*)d_in[14];
  const float* b1       = (const float*)d_in[15];
  const float* W2       = (const float*)d_in[16];
  const float* b2       = (const float*)d_in[17];
  const float* ln1_g    = (const float*)d_in[18];
  const float* ln1_b    = (const float*)d_in[19];
  const float* ln2_g    = (const float*)d_in[20];
  const float* ln2_b    = (const float*)d_in[21];
  const float* ln3_g    = (const float*)d_in[22];
  const float* ln3_b    = (const float*)d_in[23];

  // workspace carve-up
  float*  x0   = (float*)d_ws;                       // 64x1024
  float*  x1   = x0 + BATCH * DMODEL;                // 64x1024
  float*  qb   = x1 + BATCH * DMODEL;                // 64x1024
  float*  ctxb = qb + BATCH * DMODEL;                // 64x1024
  float*  db   = ctxb + BATCH * DMODEL;              // 64x1024
  float*  tmpA = db + BATCH * DMODEL;                // 64x4096 (qkv0 / ffn hidden)
  bf16_t* kv   = (bf16_t*)(tmpA + BATCH * DFFN);     // 32768 x 2048 bf16 (128MB)

  hipMemcpyAsync(x0, tgt_last, (size_t)BATCH * DMODEL * sizeof(float),
                 hipMemcpyDeviceToDevice, stream);

  float* xc = x0;
  float* xn = x1;
  const dim3 blk(256);

  for (int i = 0; i < LAYERS; ++i) {
    const float* saW = sa_W + (long)i * 3 * DMODEL * DMODEL;
    const float* sab = sa_b + (long)i * 3 * DMODEL;

    // ---------------- self attention ----------------
    if (i == 0) {
      // qkv for the single new token: (64, 3072)
      gemm_wmma_bf16<<<dim3(3 * DMODEL / BN, 1), blk, 0, stream>>>(
          xc, xc, BATCH, BATCH, saW, sab, DMODEL, tmpA, nullptr, 3 * DMODEL, 0);
      attn_f32<<<BATCH * NHEAD, blk, 0, stream>>>(
          tmpA, 3 * DMODEL, K0, V0, TCACHE,
          tmpA + DMODEL, tmpA + 2 * DMODEL, 3 * DMODEL, ctxb);
    } else {
      // q projection for x
      gemm_wmma_bf16<<<dim3(DMODEL / BN, 1), blk, 0, stream>>>(
          xc, xc, BATCH, BATCH, saW, sab, DMODEL, qb, nullptr, DMODEL, 0);
      // K|V projection of concat(cache[i], x): 32768 x 2048, bf16 out
      const long M0 = (long)TCACHE * BATCH;          // 32704
      const long Mkv = (long)(TCACHE + 1) * BATCH;   // 32768
      gemm_wmma_bf16<<<dim3(2 * DMODEL / BN, Mkv / BM), blk, 0, stream>>>(
          cache + (long)i * TCACHE * BATCH * DMODEL, xc, M0, Mkv,
          saW + (long)DMODEL * DMODEL, sab + DMODEL, DMODEL,
          nullptr, kv, 2 * DMODEL, 0);
      attn_kv_bf16<<<BATCH * NHEAD, blk, 0, stream>>>(qb, kv, ctxb);
    }
    gemm_wmma_bf16<<<dim3(DMODEL / BN, 1), blk, 0, stream>>>(
        ctxb, ctxb, BATCH, BATCH, sa_oW + (long)i * DMODEL * DMODEL,
        sa_ob + (long)i * DMODEL, DMODEL, db, nullptr, DMODEL, 0);
    ln_res<<<BATCH, blk, 0, stream>>>(xc, db, ln1_g + (long)i * DMODEL,
                                      ln1_b + (long)i * DMODEL, xn);
    { float* t = xc; xc = xn; xn = t; }

    // ---------------- cross attention ----------------
    gemm_wmma_bf16<<<dim3(DMODEL / BN, 1), blk, 0, stream>>>(
        xc, xc, BATCH, BATCH, ca_qW + (long)i * DMODEL * DMODEL,
        ca_qb + (long)i * DMODEL, DMODEL, qb, nullptr, DMODEL, 0);
    attn_f32<<<BATCH * NHEAD, blk, 0, stream>>>(
        qb, DMODEL,
        K_mem + (long)i * BATCH * NHEAD * SMEMT * DHEAD,
        V_mem + (long)i * BATCH * NHEAD * SMEMT * DHEAD,
        SMEMT, nullptr, nullptr, 0, ctxb);
    gemm_wmma_bf16<<<dim3(DMODEL / BN, 1), blk, 0, stream>>>(
        ctxb, ctxb, BATCH, BATCH, ca_oW + (long)i * DMODEL * DMODEL,
        ca_ob + (long)i * DMODEL, DMODEL, db, nullptr, DMODEL, 0);
    ln_res<<<BATCH, blk, 0, stream>>>(xc, db, ln2_g + (long)i * DMODEL,
                                      ln2_b + (long)i * DMODEL, xn);
    { float* t = xc; xc = xn; xn = t; }

    // ---------------- FFN ----------------
    gemm_wmma_bf16<<<dim3(DFFN / BN, 1), blk, 0, stream>>>(
        xc, xc, BATCH, BATCH, W1 + (long)i * DFFN * DMODEL,
        b1 + (long)i * DFFN, DMODEL, tmpA, nullptr, DFFN, 1 /*relu*/);
    gemm_wmma_bf16<<<dim3(DMODEL / BN, 1), blk, 0, stream>>>(
        tmpA, tmpA, BATCH, BATCH, W2 + (long)i * DMODEL * DFFN,
        b2 + (long)i * DMODEL, DFFN, db, nullptr, DMODEL, 0);
    ln_res<<<BATCH, blk, 0, stream>>>(xc, db, ln3_g + (long)i * DMODEL,
                                      ln3_b + (long)i * DMODEL, xn);
    { float* t = xc; xc = xn; xn = t; }
  }

  hipMemcpyAsync(d_out, xc, (size_t)BATCH * DMODEL * sizeof(float),
                 hipMemcpyDeviceToDevice, stream);
}
